// SchNet_18502719111263
// MI455X (gfx1250) — matrix operand order repaired and checked by reading, not compiled, not run on hardware
//
#include <hip/hip_runtime.h>
#include <hip/hip_bf16.h>
#include <stdint.h>

typedef __attribute__((ext_vector_type(16))) _Float16 v16h;
typedef __attribute__((ext_vector_type(8)))  _Float16 v8h;
typedef __attribute__((ext_vector_type(8)))  float    v8f;

#define NNODES  100000
#define NEDGES  2000000
#define NCENT   50
#define RBF_GAP (5.0f/49.0f)
#define EDGE_ITERS 5              // 5 x 128 edges per 256-thread block -> grid 3125

// ---- WMMA fragment loaders (CDNA5 16-bit layouts, CDNA5 ISA 7.12.2) ----
// A: 16x32 f16, row m = lane%16; lanes<16: elems 0..7 = K0..7, 8..15 = K16..23;
//    lanes>=16: elems 0..7 = K8..15, 8..15 = K24..31.
__device__ __forceinline__ v16h load_a_frag(const _Float16* tile, int lane, int kstep) {
    int hh = lane >> 4, m = lane & 15;
    const _Float16* row = tile + m * 64 + kstep * 32 + hh * 8;
    v8h lo = *(const v8h*)(row);
    v8h hi = *(const v8h*)(row + 16);
    v16h a;
#pragma unroll
    for (int j = 0; j < 8; ++j) { a[j] = lo[j]; a[8 + j] = hi[j]; }
    return a;
}

// B: 32x16 f16, B[k][n] = W[n][k] with W row-major [64][64] f16.
// lanes<16: elems 0..7 = K0..7, 8..15 = K16..23; lanes>=16: elems 0..7 = K8..15, 8..15 = K24..31.
__device__ __forceinline__ v16h load_b_frag(const _Float16* W, int lane, int ntile, int kstep) {
    int hh = lane >> 4, n = lane & 15;
    const _Float16* p = W + (ntile * 16 + n) * 64 + kstep * 32 + hh * 8;
    v8h lo = *(const v8h*)(p);
    v8h hi = *(const v8h*)(p + 16);
    v16h b;
#pragma unroll
    for (int j = 0; j < 8; ++j) { b[j] = lo[j]; b[8 + j] = hi[j]; }
    return b;
}

__device__ __forceinline__ float softplus_b05(float x) {   // beta=0.5, threshold=14
    float y = 0.5f * x;
    return (y > 14.0f) ? x : 2.0f * __logf(1.0f + __expf(y));
}
__device__ __forceinline__ float softplus_b1(float x) {    // beta=1, threshold=20
    return (x > 20.0f) ? x : __logf(1.0f + __expf(x));
}

// ---------------------------------------------------------------------------
__global__ void embed_kernel(const int* __restrict__ types, const float* __restrict__ emb,
                             float* __restrict__ h) {
    int i = blockIdx.x * 256 + threadIdx.x;     // over N*64
    int node = i >> 6, d = i & 63;
    h[i] = emb[types[node] * 64 + d];
}

__global__ void zero_kernel(float* __restrict__ p, long n) {
    long i = (long)blockIdx.x * 256 + threadIdx.x;
    if (i < n) p[i] = 0.0f;
}

// Convert all weights to f16 into wf16[16][64][64]; cfW1 padded K 50->64 with zeros.
// slot = l*5 + {0:Wn1, 1:cfW1, 2:cfW2, 3:Wn2, 4:Wn3}, slot 15 = up_W1.
__global__ void convert_weights_kernel(const float* __restrict__ Wn1, const float* __restrict__ cfW1,
                                       const float* __restrict__ cfW2, const float* __restrict__ Wn2,
                                       const float* __restrict__ Wn3, const float* __restrict__ upW1,
                                       _Float16* __restrict__ wf16) {
    int i = blockIdx.x * 256 + threadIdx.x;     // 16*4096
    int mat = i >> 12, elem = i & 4095, n = elem >> 6, k = elem & 63;
    float v;
    if (mat == 15) v = upW1[elem];
    else {
        int l = mat / 5, m = mat - l * 5;
        if      (m == 0) v = Wn1[l * 4096 + elem];
        else if (m == 1) v = (k < NCENT) ? cfW1[l * 64 * NCENT + n * NCENT + k] : 0.0f;
        else if (m == 2) v = cfW2[l * 4096 + elem];
        else if (m == 3) v = Wn2[l * 4096 + elem];
        else             v = Wn3[l * 4096 + elem];
    }
    wf16[i] = (_Float16)v;
}

// new = h @ Wn1^T  (no bias). 128 threads / 4 waves, each wave one 16x16 n-tile.
__global__ void node_gemm_kernel(const float* __restrict__ h, const _Float16* __restrict__ W,
                                 float* __restrict__ out) {
    __shared__ _Float16 tile[16 * 64];
    int tid = threadIdx.x;
    long rowbase = (long)blockIdx.x * 16;
#pragma unroll
    for (int j = 0; j < 8; ++j) {
        int idx = tid + j * 128;
        tile[idx] = (_Float16)h[rowbase * 64 + idx];
    }
    __syncthreads();
    int wave = tid >> 5, lane = tid & 31, hh = lane >> 4, n = lane & 15;
    v16h a0 = load_a_frag(tile, lane, 0);
    v16h a1 = load_a_frag(tile, lane, 1);
    v8f c = {};
    c = __builtin_amdgcn_wmma_f32_16x16x32_f16(false, a0, false, load_b_frag(W, lane, wave, 0), (short)0, c, false, false);
    c = __builtin_amdgcn_wmma_f32_16x16x32_f16(false, a1, false, load_b_frag(W, lane, wave, 1), (short)0, c, false, false);
    int col = wave * 16 + n;
#pragma unroll
    for (int r = 0; r < 8; ++r) {
        int m = hh * 8 + r;
        out[(rowbase + m) * 64 + col] = c[r];
    }
}

// Fused edge pipeline: rbf -> softplus(rbf@cfW1^T+b1) -> @cfW2^T+b2 -> *new[src] -> atomic agg[dst].
// 256 threads = 8 waves; each wave owns 16 edges/iter and a private 16x64 f16 LDS tile.
// Both filter weight matrices staged once per block in LDS (B-frags via DS pipe, not VMEM),
// amortized over EDGE_ITERS tiles; VMEM pipe is reserved for the gather + scatter-atomics.
__global__ void edge_kernel(const float* __restrict__ dist, const int* __restrict__ src,
                            const int* __restrict__ dst,
                            const _Float16* __restrict__ cfW1, const float* __restrict__ cfb1,
                            const _Float16* __restrict__ cfW2, const float* __restrict__ cfb2,
                            const float* __restrict__ newbuf, float* __restrict__ agg) {
    __shared__ _Float16 wlds[2 * 4096];          // [0]=cfW1, [4096]=cfW2 (f16, 16KB)
    __shared__ _Float16 tile[8][16 * 64];        // per-wave 16x64 tile (16KB)
    __shared__ int sidx[8][32];                  // [0..15]=src, [16..31]=dst
    int tid = threadIdx.x;
    int wave = tid >> 5, lane = tid & 31, hh = lane >> 4, n = lane & 15;

    // stage both weight matrices into LDS (4096 dwords, 16 per thread)
    {
        const uint32_t* s1 = (const uint32_t*)cfW1;
        const uint32_t* s2 = (const uint32_t*)cfW2;
        uint32_t* wd = (uint32_t*)wlds;
#pragma unroll
        for (int j = 0; j < 16; ++j) {
            int idx = tid + j * 256;
            wd[idx] = (idx < 2048) ? s1[idx] : s2[idx - 2048];
        }
    }
    __syncthreads();

    int base = blockIdx.x * (128 * EDGE_ITERS);
    for (int it = 0; it < EDGE_ITERS; ++it) {
        int ebase = base + it * 128 + wave * 16;

        // prefetch next iteration's streams (global_prefetch_b8)
        if (it + 1 < EDGE_ITERS && lane == 0) {
            int enext = ebase + 128;
            __builtin_prefetch(dist + enext, 0, 0);
            __builtin_prefetch(src + enext, 0, 0);
            __builtin_prefetch(dst + enext, 0, 0);
        }

        // stage RBF (K padded 50->64 with zeros): lane handles row n, k-half hh
        {
            float d = dist[ebase + n];
            int k0 = hh * 32;
#pragma unroll
            for (int kk = 0; kk < 32; ++kk) {
                int k = k0 + kk;
                float v = 0.0f;
                if (k < NCENT) {
                    float t = d - (float)k * RBF_GAP;
                    v = __expf(-t * t * (1.0f / RBF_GAP));
                }
                tile[wave][n * 64 + k] = (_Float16)v;
            }
            if (hh == 0) sidx[wave][n]      = src[ebase + n];
            else         sidx[wave][16 + n] = dst[ebase + n];
        }
        __syncthreads();

        // GEMM1: u = rbf @ cfW1^T   (K=64 padded, 2 ksteps x 4 ntiles, B from LDS)
        v16h a0 = load_a_frag(tile[wave], lane, 0);
        v16h a1 = load_a_frag(tile[wave], lane, 1);
        v8f c[4];
#pragma unroll
        for (int t = 0; t < 4; ++t) {
            v8f acc = {};
            acc = __builtin_amdgcn_wmma_f32_16x16x32_f16(false, a0, false, load_b_frag(wlds, lane, t, 0), (short)0, acc, false, false);
            acc = __builtin_amdgcn_wmma_f32_16x16x32_f16(false, a1, false, load_b_frag(wlds, lane, t, 1), (short)0, acc, false, false);
            c[t] = acc;
        }
        __syncthreads();
        // softplus + bias, write u back to the tile as f16
#pragma unroll
        for (int t = 0; t < 4; ++t) {
            float bv = cfb1[t * 16 + n];
#pragma unroll
            for (int r = 0; r < 8; ++r) {
                int m = hh * 8 + r;
                tile[wave][m * 64 + t * 16 + n] = (_Float16)softplus_b05(c[t][r] + bv);
            }
        }
        __syncthreads();

        // GEMM2: f = u @ cfW2^T + b2   (B from LDS)
        a0 = load_a_frag(tile[wave], lane, 0);
        a1 = load_a_frag(tile[wave], lane, 1);
#pragma unroll
        for (int t = 0; t < 4; ++t) {
            v8f acc = {};
            acc = __builtin_amdgcn_wmma_f32_16x16x32_f16(false, a0, false, load_b_frag(wlds + 4096, lane, t, 0), (short)0, acc, false, false);
            acc = __builtin_amdgcn_wmma_f32_16x16x32_f16(false, a1, false, load_b_frag(wlds + 4096, lane, t, 1), (short)0, acc, false, false);
            c[t] = acc;
        }

        // message = new[src] * f ; segment_sum via atomics on agg[dst] (L2-resident tables)
        // half-wave coalesced: for fixed (t,r), lanes n=0..15 touch 16 consecutive dwords.
#pragma unroll
        for (int t = 0; t < 4; ++t) {
            float bv = cfb2[t * 16 + n];
            int col = t * 16 + n;
#pragma unroll
            for (int r = 0; r < 8; ++r) {
                int m = hh * 8 + r;
                int s  = sidx[wave][m];
                int dd = sidx[wave][16 + m];
                float val = (c[t][r] + bv) * newbuf[(long)s * 64 + col];
                atomicAdd(&agg[(long)dd * 64 + col], val);
            }
        }
        __syncthreads();   // tile/sidx reused next iteration
    }
}

// t = softplus(agg@Wn2^T + bn2); h += t@Wn3^T + bn3   (fused, per 16-row tile)
__global__ void node_update_kernel(const float* __restrict__ agg, const _Float16* __restrict__ Wn2,
                                   const float* __restrict__ bn2, const _Float16* __restrict__ Wn3,
                                   const float* __restrict__ bn3, float* __restrict__ h) {
    __shared__ _Float16 tile[16 * 64];
    int tid = threadIdx.x;
    long rowbase = (long)blockIdx.x * 16;
#pragma unroll
    for (int j = 0; j < 8; ++j) {
        int idx = tid + j * 128;
        tile[idx] = (_Float16)agg[rowbase * 64 + idx];
    }
    __syncthreads();
    int wave = tid >> 5, lane = tid & 31, hh = lane >> 4, n = lane & 15;
    v16h a0 = load_a_frag(tile, lane, 0);
    v16h a1 = load_a_frag(tile, lane, 1);
    v8f c = {};
    c = __builtin_amdgcn_wmma_f32_16x16x32_f16(false, a0, false, load_b_frag(Wn2, lane, wave, 0), (short)0, c, false, false);
    c = __builtin_amdgcn_wmma_f32_16x16x32_f16(false, a1, false, load_b_frag(Wn2, lane, wave, 1), (short)0, c, false, false);
    float bv = bn2[wave * 16 + n];
    __syncthreads();
#pragma unroll
    for (int r = 0; r < 8; ++r) {
        int m = hh * 8 + r;
        tile[m * 64 + wave * 16 + n] = (_Float16)softplus_b05(c[r] + bv);
    }
    __syncthreads();
    a0 = load_a_frag(tile, lane, 0);
    a1 = load_a_frag(tile, lane, 1);
    v8f c2 = {};
    c2 = __builtin_amdgcn_wmma_f32_16x16x32_f16(false, a0, false, load_b_frag(Wn3, lane, wave, 0), (short)0, c2, false, false);
    c2 = __builtin_amdgcn_wmma_f32_16x16x32_f16(false, a1, false, load_b_frag(Wn3, lane, wave, 1), (short)0, c2, false, false);
    float bv3 = bn3[wave * 16 + n];
    int col = wave * 16 + n;
#pragma unroll
    for (int r = 0; r < 8; ++r) {
        int m = hh * 8 + r;
        long o = (rowbase + m) * 64 + col;
        h[o] = h[o] + c2[r] + bv3;
    }
}

// u = softplus1(h@upW1^T + b1) - ln2 ; scalar = u.upW2 + b2 ; atomic out[graph] += scalar
__global__ void readout_kernel(const float* __restrict__ h, const _Float16* __restrict__ W1f,
                               const float* __restrict__ b1, const float* __restrict__ W2,
                               const float* __restrict__ b2, const int* __restrict__ gid,
                               float* __restrict__ out) {
    __shared__ _Float16 tile[16 * 64];
    __shared__ float u[16 * 64];
    int tid = threadIdx.x;
    long rowbase = (long)blockIdx.x * 16;
#pragma unroll
    for (int j = 0; j < 8; ++j) {
        int idx = tid + j * 128;
        tile[idx] = (_Float16)h[rowbase * 64 + idx];
    }
    __syncthreads();
    int wave = tid >> 5, lane = tid & 31, hh = lane >> 4, n = lane & 15;
    v16h a0 = load_a_frag(tile, lane, 0);
    v16h a1 = load_a_frag(tile, lane, 1);
    v8f c = {};
    c = __builtin_amdgcn_wmma_f32_16x16x32_f16(false, a0, false, load_b_frag(W1f, lane, wave, 0), (short)0, c, false, false);
    c = __builtin_amdgcn_wmma_f32_16x16x32_f16(false, a1, false, load_b_frag(W1f, lane, wave, 1), (short)0, c, false, false);
    float bv = b1[wave * 16 + n];
#pragma unroll
    for (int r = 0; r < 8; ++r) {
        int m = hh * 8 + r;
        u[m * 64 + wave * 16 + n] = softplus_b1(c[r] + bv) - 0.69314718056f;
    }
    __syncthreads();
    if (tid < 16) {
        float acc = b2[0];
#pragma unroll
        for (int k = 0; k < 64; ++k) acc += u[tid * 64 + k] * W2[k];
        atomicAdd(&out[gid[rowbase + tid]], acc);
    }
}

// ---------------------------------------------------------------------------
extern "C" void kernel_launch(void* const* d_in, const int* in_sizes, int n_in,
                              void* d_out, int out_size, void* d_ws, size_t ws_size,
                              hipStream_t stream) {
    (void)in_sizes; (void)n_in; (void)ws_size;
    const int*   atom_types = (const int*)d_in[0];
    const float* dist  = (const float*)d_in[1];
    const int*   src   = (const int*)d_in[2];
    const int*   dst   = (const int*)d_in[3];
    const int*   gid   = (const int*)d_in[4];
    const float* emb   = (const float*)d_in[6];
    const float* Wn1   = (const float*)d_in[7];
    const float* cfW1  = (const float*)d_in[8];
    const float* cfb1  = (const float*)d_in[9];
    const float* cfW2  = (const float*)d_in[10];
    const float* cfb2  = (const float*)d_in[11];
    const float* Wn2   = (const float*)d_in[12];
    const float* bn2   = (const float*)d_in[13];
    const float* Wn3   = (const float*)d_in[14];
    const float* bn3   = (const float*)d_in[15];
    const float* upW1  = (const float*)d_in[16];
    const float* upb1  = (const float*)d_in[17];
    const float* upW2  = (const float*)d_in[18];
    const float* upb2  = (const float*)d_in[19];

    float* h    = (float*)d_ws;                      // N*64 f32
    float* nbuf = h    + (size_t)NNODES * 64;        // N*64 f32
    float* agg  = nbuf + (size_t)NNODES * 64;        // N*64 f32
    _Float16* wf16 = (_Float16*)(agg + (size_t)NNODES * 64);   // 16*4096 f16

    embed_kernel<<<NNODES * 64 / 256, 256, 0, stream>>>(atom_types, emb, h);
    convert_weights_kernel<<<256, 256, 0, stream>>>(Wn1, cfW1, cfW2, Wn2, Wn3, upW1, wf16);
    zero_kernel<<<(out_size + 255) / 256, 256, 0, stream>>>((float*)d_out, (long)out_size);

    for (int l = 0; l < 3; ++l) {
        const _Float16* wWn1 = wf16 + (l * 5 + 0) * 4096;
        const _Float16* wcf1 = wf16 + (l * 5 + 1) * 4096;
        const _Float16* wcf2 = wf16 + (l * 5 + 2) * 4096;
        const _Float16* wWn2 = wf16 + (l * 5 + 3) * 4096;
        const _Float16* wWn3 = wf16 + (l * 5 + 4) * 4096;
        zero_kernel<<<NNODES * 64 / 256, 256, 0, stream>>>(agg, (long)NNODES * 64);
        node_gemm_kernel<<<NNODES / 16, 128, 0, stream>>>(h, wWn1, nbuf);
        edge_kernel<<<NEDGES / (128 * EDGE_ITERS), 256, 0, stream>>>(dist, src, dst,
                                                      wcf1, cfb1 + l * 64,
                                                      wcf2, cfb2 + l * 64, nbuf, agg);
        node_update_kernel<<<NNODES / 16, 128, 0, stream>>>(agg, wWn2, bn2 + l * 64,
                                                            wWn3, bn3 + l * 64, h);
    }
    readout_kernel<<<NNODES / 16, 128, 0, stream>>>(h, wf16 + 15 * 4096, upb1, upW2, upb2,
                                                    gid, (float*)d_out);
}
